// QuaternionLinear_32744830665498
// MI455X (gfx1250) — matrix operand order repaired and checked by reading, not compiled, not run on hardware
//
#include <hip/hip_runtime.h>
#include <hip/hip_bf16.h>

// ---- CDNA5 WMMA vector types ----
typedef __attribute__((ext_vector_type(16))) __bf16 v16bf;
typedef __attribute__((ext_vector_type(8)))  __bf16 v8bf;
typedef __attribute__((ext_vector_type(8)))  float  v8f;

// Problem constants (match reference)
#define IN_F   2048
#define OUT_F  2048
#define IQn    512     // IN_F/4 quaternion input channels
#define OQn    512     // OUT_F/4 quaternion output channels
#define MROWS  16384   // B*S

// Tile: 64 rows x 32 output-quaternions per block; K chunk = 32 quaternions.
// 8 waves: wave = (rg in 0..3) x (cg in 0..1); each wave owns a 16x16 tile
// per product => 10 v8f accumulators (Ar,Br,Cr,Dr,Ai,Bi,Aj,Cj,Ak,Dk).

__device__ __forceinline__ v8f wmma_bf16(v16bf a, v16bf b, v8f c) {
    return __builtin_amdgcn_wmma_f32_16x16x32_bf16(
        /*neg_a=*/false, a, /*neg_b=*/false, b,
        /*c_mod=*/(short)0, c, /*reuse_a=*/false, /*reuse_b=*/false);
}

__global__ __launch_bounds__(256)
void qlin_wmma_kernel(const float* __restrict__ x,
                      const float* __restrict__ wr, const float* __restrict__ wi,
                      const float* __restrict__ wj, const float* __restrict__ wk,
                      const float* __restrict__ bias,
                      float* __restrict__ out)
{
    // Fragment-major LDS tiles (contiguous k order k = 16*(lane/16) + elem, 32B contiguous per lane):
    //  A (16x32 bf16): lane = m + (k&16? 16:0), elem idx = (k&15)
    //  B (32x16 bf16): lane = n + (k&16? 16:0), elem idx = (k&15)
    __shared__ __align__(32) __bf16 As[4][4][32][16]; // [comp][rowgroup][lane][elem]
    __shared__ __align__(32) __bf16 Bs[4][2][32][16]; // [comp][colgroup][lane][elem]

    const int tid  = threadIdx.x;
    const int lane = tid & 31;
    const int wave = tid >> 5;
    const int rg   = wave >> 1;   // 0..3  row tile within block
    const int cg   = wave & 1;    // 0..1  oq  tile within block

    const int rowbase = blockIdx.y * 64;
    const int oqbase  = blockIdx.x * 32;

    // ---- per-thread staging geometry (loop-invariant) ----
    // X: each thread owns 8 CONSECUTIVE quaternions -> contiguous fragment bytes
    const int xr_   = tid >> 2;              // local row 0..63
    const int xrgrp = xr_ >> 4;
    const int xm    = xr_ & 15;
    const int qbase = (tid & 3) * 8;         // 0,8,16,24
    const int xl    = xm + ((qbase & 16) ? 16 : 0);
    const int xidx  = (qbase & 8) ? 8 : 0;
    const float* xp0 = x + (size_t)(rowbase + xr_) * IN_F + (size_t)qbase * 4;

    // W: each thread owns 16 consecutive iq of one (comp, oq) row
    const int comp  = tid >> 6;              // 0..3
    const int t2    = tid & 63;
    const int wn    = t2 >> 1;               // local oq 0..31
    const int khalf = t2 & 1;                // 0: k=0..15, 1: k=16..31
    const float* wsel = (comp == 0) ? wr : (comp == 1) ? wi : (comp == 2) ? wj : wk;
    const float* wp0  = wsel + (size_t)(oqbase + wn) * IQn + khalf * 16;
    const int wcgrp = wn >> 4;
    const int wlB   = (wn & 15) + (khalf ? 16 : 0);

    v8f accAr = {}, accBr = {}, accCr = {}, accDr = {};
    v8f accAi = {}, accBi = {}, accAj = {}, accCj = {};
    v8f accAk = {}, accDk = {};

    for (int kb = 0; kb < IQn; kb += 32) {
        // ---------- stage X tile: 8 consecutive quaternions = 128B contiguous ----------
        {
            const float* xp = xp0 + (size_t)kb * 4;
            // speculative prefetch of next K tile (dropped silently if OOB)
            __builtin_prefetch(xp + 128, 0, 3);
            float4 v[8];
            #pragma unroll
            for (int i = 0; i < 8; ++i) v[i] = ((const float4*)xp)[i];
            v8bf pr, pi, pj, pk;
            #pragma unroll
            for (int i = 0; i < 8; ++i) {
                pr[i] = (__bf16)v[i].x;
                pi[i] = (__bf16)v[i].y;
                pj[i] = (__bf16)v[i].z;
                pk[i] = (__bf16)v[i].w;
            }
            *(v8bf*)&As[0][xrgrp][xl][xidx] = pr;   // 16B ds stores
            *(v8bf*)&As[1][xrgrp][xl][xidx] = pi;
            *(v8bf*)&As[2][xrgrp][xl][xidx] = pj;
            *(v8bf*)&As[3][xrgrp][xl][xidx] = pk;
        }
        // ---------- stage W tiles: 16 consecutive iq = 32B contiguous ----------
        {
            const float* wp = wp0 + kb;
            float f[16];
            #pragma unroll
            for (int i = 0; i < 4; ++i) {
                float4 t = ((const float4*)wp)[i];
                f[4 * i + 0] = t.x; f[4 * i + 1] = t.y;
                f[4 * i + 2] = t.z; f[4 * i + 3] = t.w;
            }
            v16bf pb;
            #pragma unroll
            for (int i = 0; i < 16; ++i) pb[i] = (__bf16)f[i];
            *(v16bf*)&Bs[comp][wcgrp][wlB][0] = pb; // 32B ds store
        }
        __syncthreads();

        // ---------- fragments: one 32B contiguous LDS read each ----------
        v16bf a_r = *(const v16bf*)&As[0][rg][lane][0];
        v16bf a_i = *(const v16bf*)&As[1][rg][lane][0];
        v16bf a_j = *(const v16bf*)&As[2][rg][lane][0];
        v16bf a_k = *(const v16bf*)&As[3][rg][lane][0];
        v16bf b_r = *(const v16bf*)&Bs[0][cg][lane][0];
        v16bf b_i = *(const v16bf*)&Bs[1][cg][lane][0];
        v16bf b_j = *(const v16bf*)&Bs[2][cg][lane][0];
        v16bf b_k = *(const v16bf*)&Bs[3][cg][lane][0];

        // ---------- 10 structured products (saves 6/16 vs densified GEMM) ----------
        accAr = wmma_bf16(a_r, b_r, accAr);
        accBr = wmma_bf16(a_i, b_r, accBr);
        accCr = wmma_bf16(a_j, b_r, accCr);
        accDr = wmma_bf16(a_k, b_r, accDr);
        accAi = wmma_bf16(a_r, b_i, accAi);
        accBi = wmma_bf16(a_i, b_i, accBi);
        accAj = wmma_bf16(a_r, b_j, accAj);
        accCj = wmma_bf16(a_j, b_j, accCj);
        accAk = wmma_bf16(a_r, b_k, accAk);
        accDk = wmma_bf16(a_k, b_k, accDk);

        __syncthreads();
    }

    // ---------- epilogue: recombine with reference signs, add bias ----------
    // C/D layout: elem v, lane -> (m = v + (lane>=16 ? 8:0), n = lane&15)
    const int mhi = (lane >> 4) ? 8 : 0;
    const int n   = lane & 15;
    const int oq  = oqbase + cg * 16 + n;
    const float4 b4 = *(const float4*)&bias[4 * oq];

    #pragma unroll
    for (int v = 0; v < 8; ++v) {
        const int row = rowbase + rg * 16 + v + mhi;
        const float Ar = accAr[v], Br = accBr[v], Cr = accCr[v], Dr = accDr[v];
        const float Ai = accAi[v], Bi = accBi[v], Aj = accAj[v], Cj = accCj[v];
        const float Ak = accAk[v], Dk = accDk[v];
        float4 o;
        o.x = (Ar - Bi - Cj - Dk) + b4.x;   // or
        o.y = (Br + Ai + Dk - Cj) + b4.y;   // oi
        o.z = (Cr + Aj + Bi - Dk) + b4.z;   // oj
        o.w = (Dr + Ak + Cj - Bi) + b4.w;   // ok
        *(float4*)&out[(size_t)row * OUT_F + 4 * oq] = o;  // interleaved scatter = contiguous f4
    }
}

extern "C" void kernel_launch(void* const* d_in, const int* in_sizes, int n_in,
                              void* d_out, int out_size, void* d_ws, size_t ws_size,
                              hipStream_t stream) {
    (void)in_sizes; (void)n_in; (void)d_ws; (void)ws_size; (void)out_size;
    const float* x    = (const float*)d_in[0];
    const float* wr   = (const float*)d_in[1];
    const float* wi   = (const float*)d_in[2];
    const float* wj   = (const float*)d_in[3];
    const float* wk   = (const float*)d_in[4];
    const float* bias = (const float*)d_in[5];
    float* out = (float*)d_out;

    dim3 grid(OQn / 32, MROWS / 64);   // 16 x 256 blocks
    dim3 block(256);                   // 8 wave32 waves
    qlin_wmma_kernel<<<grid, block, 0, stream>>>(x, wr, wi, wj, wk, bias, out);
}
